// BahdanauAttention_77051713290520
// MI455X (gfx1250) — matrix operand order repaired and checked by reading, not compiled, not run on hardware
//
#include <hip/hip_runtime.h>
#include <hip/hip_bf16.h>
#include <math.h>

// ---------------------------------------------------------------------------
// Bahdanau attention, fused for MI455X (gfx1250, wave32, WMMA).
//   B=32, S=2048, H=U=1024
//   K1: q = dec_hidden @ W1 + b1                      (tiny)
//   K2: score[b,s] = tanh(q[b]+enc[b,s]@W2+b2) . V    (137 GFLOP bf16 WMMA;
//        A tile LDS-resident (bank-conflict-free padded), B panel staged with
//        register pipelining; A/B fragments double-buffered in the K loop)
//   K3: attn = softmax_S(score)                       (bV shift-invariant)
//   K4: ctx[b,h] = sum_s attn[b,s]*enc[b,s,h]
// ---------------------------------------------------------------------------

typedef __attribute__((ext_vector_type(16))) __bf16 v16bf;
typedef __attribute__((ext_vector_type(8)))  float  v8f;

#define B_ 32
#define S_ 2048
#define H_ 1024
#define U_ 1024

#define ASTR 1032            // ushorts per enc row in LDS: stride ≡ 4 dwords
                             // (mod 64 banks) -> 16 lanes hit disjoint banks
#define BSTR 1032            // same trick for the W2 panel
#define BPANEL (16 * BSTR)

union Frag { v16bf v; uint4 u4[2]; };

__device__ __forceinline__ unsigned f2bf(float f) {
  union { float f; unsigned u; } v; v.f = f;
  unsigned r = v.u + 0x7FFFu + ((v.u >> 16) & 1u);   // round-to-nearest-even
  return r >> 16;
}

// ---------------- K1: q = dec @ W1 + b1  ([32,1024]) -----------------------
__global__ void qproj_kernel(const float* __restrict__ dec,
                             const float* __restrict__ W1,
                             const float* __restrict__ b1,
                             float* __restrict__ q) {
  const int b = blockIdx.x >> 2;
  const int u = ((blockIdx.x & 3) << 8) + threadIdx.x;
  __shared__ float drow[H_];
  for (int h = threadIdx.x; h < H_; h += 256) drow[h] = dec[b * H_ + h];
  __syncthreads();
  float acc = b1[u];
  for (int h = 0; h < H_; ++h) acc += drow[h] * W1[h * U_ + u];
  q[b * U_ + u] = acc;
}

// ---------------- K2: fused score kernel -----------------------------------
// Block = 256 threads = 8 waves; block owns 128 enc rows (one batch).
__global__ void score_kernel(const float* __restrict__ enc,
                             const float* __restrict__ W2,
                             const float* __restrict__ b2,
                             const float* __restrict__ q,
                             const float* __restrict__ Vv,
                             float* __restrict__ score) {
  extern __shared__ unsigned short lds[];
  unsigned short* As = lds;                 // [128][ASTR] bf16 enc rows
  unsigned short* Bsh = lds + 128 * ASTR;   // [16][BSTR]  bf16 W2 panel (n-major)

  const int  tid  = threadIdx.x;
  const int  lane = tid & 31;
  const int  wave = tid >> 5;
  const long rowbase = (long)blockIdx.x * 128;   // 2048 % 128 == 0
  const int  b = (int)(rowbase >> 11);

  // ---- Stage A: 128x1024 f32 -> bf16 LDS (float4 loads, packed b32 stores) --
  for (int i = tid; i < 128 * 256; i += 256) {
    const int m  = i >> 8;
    const int k4 = (i & 255) << 2;
    const float4 f = *(const float4*)(enc + (rowbase + m) * H_ + k4);
    *(unsigned*)(As + m * ASTR + k4)     = f2bf(f.x) | (f2bf(f.y) << 16);
    *(unsigned*)(As + m * ASTR + k4 + 2) = f2bf(f.z) | (f2bf(f.w) << 16);
  }

  // ---- Stage B panel 0 (W2[:,0:16] -> Bsh[n][k], bf16, transposed) ----
  const int n4    = (tid & 3) << 2;   // this thread's 4 n-columns
  const int kbase = tid >> 2;         // + j*64 covers k = 0..1023
#pragma unroll 4
  for (int j = 0; j < 16; ++j) {
    const int k = kbase + j * 64;
    const float4 f = *(const float4*)(W2 + k * U_ + n4);
    Bsh[(n4 + 0) * BSTR + k] = (unsigned short)f2bf(f.x);
    Bsh[(n4 + 1) * BSTR + k] = (unsigned short)f2bf(f.y);
    Bsh[(n4 + 2) * BSTR + k] = (unsigned short)f2bf(f.z);
    Bsh[(n4 + 3) * BSTR + k] = (unsigned short)f2bf(f.w);
  }
  __syncthreads();

  // WMMA wave32 operand layout (ISA 7.12.2, 16-bit):
  //   A lane L: row M=L%16, K chunks {0..7,16..23} (L<16) / {8..15,24..31}
  //   B lane L: col N=L%16, K chunks {0..7,16..23} (L<16) / {8..15,24..31} (n-major LDS)
  //   C lane L: col N=L%16, rows r (L<16) / r+8 in acc[r]
  const int colN = lane & 15;
  const int half = lane >> 4;
  const unsigned short* Achunk = As + (wave * 16 + colN) * ASTR + half * 8;
  const unsigned short* Brow   = Bsh + colN * BSTR + half * 8;

  float part[8] = {0.f, 0.f, 0.f, 0.f, 0.f, 0.f, 0.f, 0.f};

  for (int nt = 0; nt < 64; ++nt) {
    const int n0 = nt << 4;

    // Phase 1: issue global loads for panel nt+1 (hide under WMMA phase)
    float4 wreg[16];
    if (nt + 1 < 64) {
#pragma unroll
      for (int j = 0; j < 16; ++j) {
        const int k = kbase + j * 64;
        wreg[j] = *(const float4*)(W2 + k * U_ + (n0 + 16) + n4);
      }
    }

    // Phase 2: K sweep, 32 WMMAs, A/B fragments double-buffered in registers
    v8f acc = {};
    {
      Frag fa0, fa1, fb0, fb1;
      fa0.u4[0] = *(const uint4*)(Achunk);
      fa0.u4[1] = *(const uint4*)(Achunk + 16);
      fb0.u4[0] = *(const uint4*)(Brow);
      fb0.u4[1] = *(const uint4*)(Brow + 16);
#pragma unroll
      for (int ks = 0; ks < 32; ks += 2) {
        fa1.u4[0] = *(const uint4*)(Achunk + (ks + 1) * 32);
        fa1.u4[1] = *(const uint4*)(Achunk + (ks + 1) * 32 + 16);
        fb1.u4[0] = *(const uint4*)(Brow + (ks + 1) * 32);
        fb1.u4[1] = *(const uint4*)(Brow + (ks + 1) * 32 + 16);
        acc = __builtin_amdgcn_wmma_f32_16x16x32_bf16(
                  false, fa0.v, false, fb0.v, (short)0, acc, false, false);
        if (ks + 2 < 32) {
          fa0.u4[0] = *(const uint4*)(Achunk + (ks + 2) * 32);
          fa0.u4[1] = *(const uint4*)(Achunk + (ks + 2) * 32 + 16);
          fb0.u4[0] = *(const uint4*)(Brow + (ks + 2) * 32);
          fb0.u4[1] = *(const uint4*)(Brow + (ks + 2) * 32 + 16);
        }
        acc = __builtin_amdgcn_wmma_f32_16x16x32_bf16(
                  false, fa1.v, false, fb1.v, (short)0, acc, false, false);
      }
    }

    // Epilogue: tanh(k + q + b2) * V, accumulate per-row partials
    const int   n  = n0 + colN;
    const float qb = q[b * U_ + n] + b2[n];
    const float vn = Vv[n];
#pragma unroll
    for (int r = 0; r < 8; ++r)
      part[r] += tanhf(acc[r] + qb) * vn;

    // Phase 3: all waves done reading the panel -> overwrite with nt+1
    __syncthreads();
    if (nt + 1 < 64) {
#pragma unroll
      for (int j = 0; j < 16; ++j) {
        const int k = kbase + j * 64;
        Bsh[(n4 + 0) * BSTR + k] = (unsigned short)f2bf(wreg[j].x);
        Bsh[(n4 + 1) * BSTR + k] = (unsigned short)f2bf(wreg[j].y);
        Bsh[(n4 + 2) * BSTR + k] = (unsigned short)f2bf(wreg[j].z);
        Bsh[(n4 + 3) * BSTR + k] = (unsigned short)f2bf(wreg[j].w);
      }
      __syncthreads();
    }
  }

  // Reduce the 16 columns held across lanes of each half-wave
#pragma unroll
  for (int off = 1; off < 16; off <<= 1)
#pragma unroll
    for (int r = 0; r < 8; ++r)
      part[r] += __shfl_xor(part[r], off, 32);
  if ((lane & 15) == 0) {
    const long base = rowbase + wave * 16 + half * 8;
#pragma unroll
    for (int r = 0; r < 8; ++r) score[base + r] = part[r];
  }
}

// ---------------- K3: softmax over S per batch -----------------------------
__global__ void softmax_kernel(const float* __restrict__ score,
                               float* __restrict__ attn) {
  const int b = blockIdx.x;
  const float* s = score + b * S_;
  __shared__ float red[256];
  float m = -1e30f;
  for (int i = threadIdx.x; i < S_; i += 256) m = fmaxf(m, s[i]);
  red[threadIdx.x] = m; __syncthreads();
  for (int o = 128; o > 0; o >>= 1) {
    if (threadIdx.x < o) red[threadIdx.x] = fmaxf(red[threadIdx.x], red[threadIdx.x + o]);
    __syncthreads();
  }
  m = red[0]; __syncthreads();
  float sum = 0.f;
  for (int i = threadIdx.x; i < S_; i += 256) sum += __expf(s[i] - m);
  red[threadIdx.x] = sum; __syncthreads();
  for (int o = 128; o > 0; o >>= 1) {
    if (threadIdx.x < o) red[threadIdx.x] += red[threadIdx.x + o];
    __syncthreads();
  }
  const float inv = 1.0f / red[0];
  for (int i = threadIdx.x; i < S_; i += 256)
    attn[b * S_ + i] = __expf(s[i] - m) * inv;
}

// ---------------- K4: ctx[b,h] = sum_s attn[b,s]*enc[b,s,h] ----------------
__global__ void context_kernel(const float* __restrict__ enc,
                               const float* __restrict__ attn,
                               float* __restrict__ ctx) {
  const int b = blockIdx.x >> 2;
  const int h = ((blockIdx.x & 3) << 8) + threadIdx.x;
  __shared__ float w[S_];
  for (int i = threadIdx.x; i < S_; i += 256) w[i] = attn[b * S_ + i];
  __syncthreads();
  float acc = 0.f;
  const float* e = enc + (long)b * S_ * H_ + h;
  for (int s = 0; s < S_; ++s) acc += w[s] * e[(long)s * H_];
  ctx[b * H_ + h] = acc;
}

// ---------------------------------------------------------------------------
extern "C" void kernel_launch(void* const* d_in, const int* in_sizes, int n_in,
                              void* d_out, int out_size, void* d_ws, size_t ws_size,
                              hipStream_t stream) {
  const float* dec = (const float*)d_in[0];
  const float* enc = (const float*)d_in[1];
  const float* W1  = (const float*)d_in[2];
  const float* b1  = (const float*)d_in[3];
  const float* W2  = (const float*)d_in[4];
  const float* b2  = (const float*)d_in[5];
  const float* Vv  = (const float*)d_in[6];
  // d_in[7] = bV: softmax is shift-invariant -> no-op.

  float* ctx   = (float*)d_out;            // [32,1024]
  float* attn  = (float*)d_out + B_ * H_;  // [32,2048,1]
  float* q     = (float*)d_ws;             // [32,1024]
  float* score = q + B_ * U_;              // [32,2048]

  qproj_kernel<<<(B_ * U_) / 256, 256, 0, stream>>>(dec, W1, b1, q);

  const size_t ldsB = (size_t)(128 * ASTR + BPANEL) * sizeof(unsigned short); // ~297 KB
  score_kernel<<<(B_ * S_) / 128, 256, ldsB, stream>>>(enc, W2, b2, q, Vv, score);

  softmax_kernel<<<B_, 256, 0, stream>>>(score, attn);
  context_kernel<<<B_ * 4, 256, 0, stream>>>(enc, attn, ctx);
}
